// VectorQuantizer_16475494548012
// MI455X (gfx1250) — matrix operand order repaired and checked by reading, not compiled, not run on hardware
//
#include <hip/hip_runtime.h>
#include <stdint.h>

#define N_PTS 16384
#define K_CB  8192
#define D_DIM 512
#define CTILES (K_CB / 16)        // 512 codebook column tiles total
#define SPLITK 2                  // blocks per row-tile scanning disjoint K halves
#define CT_SPLIT (CTILES / SPLITK)

typedef __attribute__((ext_vector_type(16))) __bf16 v16bf;
typedef __attribute__((ext_vector_type(8)))  float  v8f;
typedef int v4i_gcc __attribute__((vector_size(16)));   // matches builtin param type

union V16U { v16bf v; uint4 q[2]; };

#define AS1 __attribute__((address_space(1)))
#define AS3 __attribute__((address_space(3)))

#if __has_builtin(__builtin_amdgcn_global_load_async_to_lds_b128)
#define HAVE_ASYNC_LDS 1
#else
#define HAVE_ASYNC_LDS 0
#endif

__device__ __forceinline__ void async_wait0() {
#if __has_builtin(__builtin_amdgcn_s_wait_asynccnt)
  __builtin_amdgcn_s_wait_asynccnt(0);
#else
  asm volatile("s_wait_asynccnt 0x0" ::: "memory");
#endif
}

__device__ __forceinline__ unsigned short f2bf(float f) {
  unsigned u = __float_as_uint(f);
  u += 0x7FFFu + ((u >> 16) & 1u);   // round-to-nearest-even
  return (unsigned short)(u >> 16);
}

#define WMMA_BF16(a, b, c) \
  __builtin_amdgcn_wmma_f32_16x16x32_bf16(false, (a), false, (b), (short)0, (c), false, false)

// ---------------- kernel 1: inputs f32 -> bf16 ----------------
__global__ void vq_cvt_x(const float* __restrict__ x, uint16_t* __restrict__ xb) {
  size_t i = ((size_t)blockIdx.x * blockDim.x + threadIdx.x) * 4;
  float4 f = *(const float4*)(x + i);
  ushort4 o = make_ushort4(f2bf(f.x), f2bf(f.y), f2bf(f.z), f2bf(f.w));
  *(ushort4*)(xb + i) = o;
}

// ------- kernel 2: codebook f32 -> bf16, c_sq[k], zero loss accumulator -------
__global__ void vq_cvt_c(const float* __restrict__ cb, uint16_t* __restrict__ cbb,
                         float* __restrict__ csq, float* __restrict__ acc) {
  const int k = blockIdx.x;        // one block per codebook row
  const int t = threadIdx.x;       // 128 threads, 4 elems each
  float4 f = *(const float4*)(cb + (size_t)k * D_DIM + t * 4);
  ushort4 o = make_ushort4(f2bf(f.x), f2bf(f.y), f2bf(f.z), f2bf(f.w));
  *(ushort4*)(cbb + (size_t)k * D_DIM + t * 4) = o;
  float s = f.x * f.x + f.y * f.y + f.z * f.z + f.w * f.w;
  #pragma unroll
  for (int sh = 16; sh >= 1; sh >>= 1) s += __shfl_xor(s, sh, 32);
  __shared__ float ps[4];
  if ((t & 31) == 0) ps[t >> 5] = s;
  __syncthreads();
  if (t == 0) {
    csq[k] = ps[0] + ps[1] + ps[2] + ps[3];
    if (k == 0) *acc = 0.0f;       // re-zero every call (graph-replay safe)
  }
}

// ---------------- kernel 3: fused bf16-WMMA distance GEMM + row argmin ----------------
// Grid = (N/128) x SPLITK. Block = 256 threads = 8 waves; each wave owns 16 rows.
// Each block scans K_CB/SPLITK codebook entries and emits one (best, idx)
// candidate per row; the gather kernel resolves the SPLITK-way min.
// A (16 rows x 512 d) is register-resident; codebook tiles (16 x 512 bf16) are
// double-buffered in LDS, staged with GLOBAL_LOAD_ASYNC_TO_LDS (ASYNCcnt) when
// available; the staging of tile ct+1 overlaps the 16 WMMAs of tile ct; 4
// independent accumulators break the WMMA RAW chain.
__global__ __launch_bounds__(256, 1) void vq_argmin(
    const uint16_t* __restrict__ xb, const uint16_t* __restrict__ cbb,
    const float* __restrict__ csq,
    float* __restrict__ bestOut, int* __restrict__ idxOut) {
  // 16 rows, row stride 520 bf16 (1040 B) -> 65 uint4 per row, bank-conflict-free
  __shared__ uint16_t btile[2][16 * 520];

  const int tid  = threadIdx.x;
  const int lane = tid & 31;
  const int wave = tid >> 5;           // 0..7
  const int m    = lane & 15;          // matrix row (A) / column (B,C)
  const int h    = lane >> 4;          // half-wave select

  // cooperative staging map: thread covers chunks (tid + 256j); dst is padded
  const int srow = tid >> 6;           // base tile row (rows advance by 4 per j)
  const int scol = tid & 63;           // 16B chunk within row

  const int rowBase = blockIdx.x * 128 + wave * 16;
  const int ktBase  = blockIdx.y * CT_SPLIT;   // first codebook tile of this split
  const uint4* arow = (const uint4*)(xb + (size_t)(rowBase + m) * D_DIM);

  // Preload all 16 A operands (covers D=512, K-chunks of 32) into registers.
  // ISA A layout (16-bit 16x32): lanes 0-15 hold K {0..7,16..23}, lanes 16-31 {8..15,24..31}.
  V16U A[16];
  #pragma unroll
  for (int t = 0; t < 16; ++t) {
    A[t].q[0] = arow[t * 4 + h];
    A[t].q[1] = arow[t * 4 + 2 + h];
  }

  float best[8];
  int   bidx[8];
  #pragma unroll
  for (int v = 0; v < 8; ++v) { best[v] = 3.4e38f; bidx[v] = 0; }

#if HAVE_ASYNC_LDS
  // DMA-style staging: global -> LDS directly, tracked by ASYNCcnt.
  {
    const uint4* src = (const uint4*)(cbb + (size_t)ktBase * 16 * D_DIM);
    uint4* dst = (uint4*)btile[0];
    #pragma unroll
    for (int j = 0; j < 4; ++j)
      __builtin_amdgcn_global_load_async_to_lds_b128(
          (AS1 v4i_gcc*)(src + tid + j * 256),
          (AS3 v4i_gcc*)(dst + (srow + 4 * j) * 65 + scol), 0, 0);
    async_wait0();
  }
#else
  {
    const uint4* src = (const uint4*)(cbb + (size_t)ktBase * 16 * D_DIM);
    uint4* dst = (uint4*)btile[0];
    #pragma unroll
    for (int j = 0; j < 4; ++j)
      dst[(srow + 4 * j) * 65 + scol] = src[tid + j * 256];
  }
#endif
  __syncthreads();

  for (int ct = 0; ct < CT_SPLIT; ++ct) {
    const int kt = ktBase + ct;
    const bool more = (ct + 1 < CT_SPLIT);

#if HAVE_ASYNC_LDS
    // Issue async staging of tile ct+1 straight into the other LDS buffer
    // (its last readers finished before the barrier that ended iteration ct-1).
    if (more) {
      const uint4* src = (const uint4*)(cbb + (size_t)(kt + 1) * 16 * D_DIM);
      uint4* dst = (uint4*)btile[(ct + 1) & 1];
      #pragma unroll
      for (int j = 0; j < 4; ++j)
        __builtin_amdgcn_global_load_async_to_lds_b128(
            (AS1 v4i_gcc*)(src + tid + j * 256),
            (AS3 v4i_gcc*)(dst + (srow + 4 * j) * 65 + scol), 0, 0);
      if (ct + 2 < CT_SPLIT)   // speculative prefetch of tile kt+2 -> global_prefetch_b8
        __builtin_prefetch((const char*)(cbb + (size_t)(kt + 2) * 16 * D_DIM) + tid * 64, 0, 1);
    }
#else
    uint4 st[4];
    if (more) {
      const uint4* src = (const uint4*)(cbb + (size_t)(kt + 1) * 16 * D_DIM);
      #pragma unroll
      for (int j = 0; j < 4; ++j) st[j] = src[tid + j * 256];
      if (ct + 2 < CT_SPLIT)
        __builtin_prefetch((const char*)(cbb + (size_t)(kt + 2) * 16 * D_DIM) + tid * 64, 0, 1);
    }
#endif

    // 16 WMMAs on the current tile, 4 independent accumulator streams.
    const uint4* ldsq = (const uint4*)btile[ct & 1];
    v8f a0 = {}, a1 = {}, a2 = {}, a3 = {};
    #pragma unroll
    for (int t = 0; t < 16; t += 4) {
      // B layout (16-bit 32x16): lane n, lanes 0-15 hold K {0..7,16..23}, 16-31 hold K {8..15,24..31}
      V16U B0, B1, B2, B3;
      B0.q[0] = ldsq[m * 65 + (t    ) * 4 + h]; B0.q[1] = ldsq[m * 65 + (t    ) * 4 + 2 + h];
      B1.q[0] = ldsq[m * 65 + (t + 1) * 4 + h]; B1.q[1] = ldsq[m * 65 + (t + 1) * 4 + 2 + h];
      B2.q[0] = ldsq[m * 65 + (t + 2) * 4 + h]; B2.q[1] = ldsq[m * 65 + (t + 2) * 4 + 2 + h];
      B3.q[0] = ldsq[m * 65 + (t + 3) * 4 + h]; B3.q[1] = ldsq[m * 65 + (t + 3) * 4 + 2 + h];
      a0 = WMMA_BF16(A[t    ].v, B0.v, a0);
      a1 = WMMA_BF16(A[t + 1].v, B1.v, a1);
      a2 = WMMA_BF16(A[t + 2].v, B2.v, a2);
      a3 = WMMA_BF16(A[t + 3].v, B3.v, a3);
    }

#if !HAVE_ASYNC_LDS
    if (more) {
      uint4* dst = (uint4*)btile[(ct + 1) & 1];
      #pragma unroll
      for (int j = 0; j < 4; ++j)
        dst[(srow + 4 * j) * 65 + scol] = st[j];
    }
#endif

    // Epilogue: dist_rel = c_sq[n] - 2*score; C layout: lane = column n, VGPR v = row
    const int   n  = kt * 16 + m;
    const float cq = csq[n];
    #pragma unroll
    for (int v = 0; v < 8; ++v) {
      float sum = (a0[v] + a1[v]) + (a2[v] + a3[v]);
      float dd  = __builtin_fmaf(-2.0f, sum, cq);
      if (dd < best[v]) { best[v] = dd; bidx[v] = n; }
    }

#if HAVE_ASYNC_LDS
    if (more) async_wait0();   // this wave's async writes to buf[(ct+1)&1] landed
#endif
    __syncthreads();           // publish staging + all reads of current buffer retired
  }

  // Cross-lane argmin within each 16-lane half (lanes 0-15: rows 0-7; 16-31: rows 8-15)
  #pragma unroll
  for (int s = 1; s < 16; s <<= 1) {
    #pragma unroll
    for (int v = 0; v < 8; ++v) {
      float ov = __shfl_xor(best[v], s, 32);
      int   oi = __shfl_xor(bidx[v], s, 32);
      if (ov < best[v] || (ov == best[v] && oi < bidx[v])) { best[v] = ov; bidx[v] = oi; }
    }
  }
  if (m == 0) {
    #pragma unroll
    for (int v = 0; v < 8; ++v) {
      const int row = rowBase + h * 8 + v;
      bestOut[row * SPLITK + blockIdx.y] = best[v];
      idxOut [row * SPLITK + blockIdx.y] = bidx[v];
    }
  }
}

// -------- kernel 4: split-K merge + gather f32 codebook rows + loss partials --------
__global__ void vq_gather_loss(const float* __restrict__ x, const float* __restrict__ cb,
                               const float* __restrict__ bestVal, const int* __restrict__ bestIdx,
                               float* __restrict__ out, float* __restrict__ acc) {
  const int n = blockIdx.x;
  const int t = threadIdx.x;   // 128 threads, 4 elems each
  // Resolve split-K candidates: split 0 holds lower indices, so ties keep j0
  // (matches jnp.argmin lowest-index semantics).
  const float d0 = bestVal[n * SPLITK + 0];
  const float d1 = bestVal[n * SPLITK + 1];
  const int   j0 = bestIdx[n * SPLITK + 0];
  const int   j1 = bestIdx[n * SPLITK + 1];
  const int   j  = (d1 < d0) ? j1 : j0;

  float4 q  = *(const float4*)(cb + (size_t)j * D_DIM + t * 4);
  float4 xv = *(const float4*)(x  + (size_t)n * D_DIM + t * 4);
  *(float4*)(out + (size_t)n * D_DIM + t * 4) = q;  // STE output == quantized
  float dx = q.x - xv.x, dy = q.y - xv.y, dz = q.z - xv.z, dw = q.w - xv.w;
  float s = dx * dx + dy * dy + dz * dz + dw * dw;
  #pragma unroll
  for (int sh = 16; sh >= 1; sh >>= 1) s += __shfl_xor(s, sh, 32);
  __shared__ float ps[4];
  if ((t & 31) == 0) ps[t >> 5] = s;
  __syncthreads();
  if (t == 0) atomicAdd(acc, ps[0] + ps[1] + ps[2] + ps[3]);
}

// ---------------- kernel 5: finalize loss ----------------
__global__ void vq_final(const float* __restrict__ acc, float* __restrict__ out) {
  // loss = q_latent + 0.25*e_latent = 1.25 * mean((q - x)^2)
  out[(size_t)N_PTS * D_DIM] = 1.25f * (*acc) / (float)((size_t)N_PTS * D_DIM);
}

extern "C" void kernel_launch(void* const* d_in, const int* in_sizes, int n_in,
                              void* d_out, int out_size, void* d_ws, size_t ws_size,
                              hipStream_t stream) {
  const float* x  = (const float*)d_in[0];   // [N, D]
  const float* cb = (const float*)d_in[1];   // [K, D]
  float* out = (float*)d_out;                // [N*D] quantized_ste, then [1] loss

  // workspace layout (~25.6 MB)
  uint16_t* xb   = (uint16_t*)d_ws;                        // N*D bf16
  uint16_t* cbb  = xb + (size_t)N_PTS * D_DIM;             // K*D bf16
  float*    csq  = (float*)(cbb + (size_t)K_CB * D_DIM);   // K f32
  float*    bval = csq + K_CB;                             // N*SPLITK f32
  int*      bidx = (int*)(bval + (size_t)N_PTS * SPLITK);  // N*SPLITK i32
  float*    acc  = (float*)(bidx + (size_t)N_PTS * SPLITK);// 1 f32

  vq_cvt_x   <<<(N_PTS * D_DIM) / 1024, 256, 0, stream>>>(x, xb);
  vq_cvt_c   <<<K_CB, 128, 0, stream>>>(cb, cbb, csq, acc);
  vq_argmin  <<<dim3(N_PTS / 128, SPLITK), 256, 0, stream>>>(xb, cbb, csq, bval, bidx);
  vq_gather_loss<<<N_PTS, 128, 0, stream>>>(x, cb, bval, bidx, out, acc);
  vq_final   <<<1, 1, 0, stream>>>(acc, out);
}
